// TypeAreaMultiHeadAttention2_84628035600753
// MI455X (gfx1250) — matrix operand order repaired and checked by reading, not compiled, not run on hardware
//
#include <hip/hip_runtime.h>
#include <math.h>

// ---- problem constants (match reference) ----
constexpr int Bb  = 4;
constexpr int Ll  = 1024;
constexpr int Dd  = 256;
constexpr int Hh  = 8;
constexpr int DKk = 32;
constexpr int BL  = Bb * Ll;   // 4096
constexpr int BH  = Bb * Hh;   // 32
constexpr float NEGV = -1e9f;
constexpr float EPSV = 1e-6f;
constexpr float QSCALE = 0.08838834764831845f; // 1/sqrt(4*dk) = 1/sqrt(128)

typedef __attribute__((ext_vector_type(16))) _Float16 v16h;
typedef __attribute__((ext_vector_type(8)))  _Float16 v8h;
typedef __attribute__((ext_vector_type(8)))  float    v8f;

#define WMMA_F16(a, b, c) \
  __builtin_amdgcn_wmma_f32_16x16x32_f16(false, (a), false, (b), (short)0, (c), false, false)

// Fragment-order LDS tiles: frag[lane] is exactly the v16h one lane feeds to WMMA.
// A-frag (16x32, MxK): lane = (k>>4)*16 + m, elem e = k&15
// B-frag (32x16, KxN): lane = (k>>4)*16 + n, elem e = k&15
__device__ __forceinline__ unsigned lds_off(const void* p) {
  return (unsigned)(uintptr_t)p;   // generic LDS pointer: addr[31:0] is the LDS offset
}
__device__ __forceinline__ void async_b128(unsigned dst_lds, const void* gsrc) {
  unsigned long long ga = (unsigned long long)gsrc;
  asm volatile("global_load_async_to_lds_b128 %0, %1, off"
               :: "v"(dst_lds), "v"(ga) : "memory");
}
__device__ __forceinline__ void wait_async0() {
  asm volatile("s_wait_asynccnt 0x0" ::: "memory");
}
__device__ __forceinline__ void cvt_store8(_Float16* dst, const float* r) {
  v8h h;
#pragma unroll
  for (int j = 0; j < 8; ++j) h[j] = (_Float16)r[j];
  *(v8h*)dst = h;   // one ds_store_b128
}

// ---------------- residual = q_a+q_s+q_bh+q_bah+q_bbh+q_pan+q_oan ----------------
__global__ void residual_kernel(const float* x0, const float* x1, const float* x2,
                                const float* x3, const float* x4, const float* x5,
                                const float* x6, float* res) {
  const size_t i = (size_t)blockIdx.x * 256 + threadIdx.x;
  res[i] = x0[i] + x1[i] + x2[i] + x3[i] + x4[i] + x5[i] + x6[i];
}

// ---- projection: out = (A0@W0 + A1@W1)*scale -> f16, layout [BH][L][DK] or (vt) [BH][DK][L]
// M=4096 N=256 K=256; tile 64x64, k-step 32, double-buffered LDS, 1 barrier/step
__global__ void proj_kernel(const float* __restrict__ A0, const float* __restrict__ W0,
                            const float* __restrict__ A1, const float* __restrict__ W1,
                            _Float16* __restrict__ out, float scale, int vt) {
  __shared__ v16h Asw[2][4][32];   // [buf][mi][lane]
  __shared__ v16h Bsw[2][4][32];   // [buf][ni][lane]
  const int tid = threadIdx.x, wave = tid >> 5, lane = tid & 31;
  const int m0 = blockIdx.x * 64, n0 = blockIdx.y * 64;
  const int mi = wave >> 1, nb = (wave & 1) * 2;
  // fill roles: A -> (row, k-octet); B -> (col, k-octet); each stores one b128
  const int arow = tid >> 2, akoct = tid & 3;
  const int bcol = tid & 63, bkoct = tid >> 6;
  _Float16* adst[2]; _Float16* bdst[2];
#pragma unroll
  for (int p = 0; p < 2; ++p) {
    adst[p] = (_Float16*)&Asw[p][arow >> 4][(akoct >> 1) * 16 + (arow & 15)] + (akoct & 1) * 8;
    bdst[p] = (_Float16*)&Bsw[p][bcol >> 4][(bkoct >> 1) * 16 + (bcol & 15)] + (bkoct & 1) * 8;
  }
  auto load_step = [&](int s, float ra[8], float rb[8]) {
    const float* A = (s & 8) ? A1 : A0;
    const float* W = (s & 8) ? W1 : W0;
    const int k0 = (s & 7) * 32;
    const float4 f0 = *(const float4*)&A[(size_t)(m0 + arow) * Dd + k0 + akoct * 8];
    const float4 f1 = *(const float4*)&A[(size_t)(m0 + arow) * Dd + k0 + akoct * 8 + 4];
    ra[0] = f0.x; ra[1] = f0.y; ra[2] = f0.z; ra[3] = f0.w;
    ra[4] = f1.x; ra[5] = f1.y; ra[6] = f1.z; ra[7] = f1.w;
#pragma unroll
    for (int j = 0; j < 8; ++j)
      rb[j] = W[(size_t)(k0 + bkoct * 8 + j) * Dd + n0 + bcol];
  };
  {
    float ra[8], rb[8];
    load_step(0, ra, rb);
    cvt_store8(adst[0], ra);
    cvt_store8(bdst[0], rb);
  }
  __syncthreads();
  v8f acc0 = {}, acc1 = {};
  for (int s = 0; s < 16; ++s) {
    const int p = s & 1;
    float ra[8], rb[8];
    if (s < 15) load_step(s + 1, ra, rb);        // next step's globals in flight
    const v16h af = Asw[p][mi][lane];
    acc0 = WMMA_F16(af, Bsw[p][nb][lane], acc0);
    acc1 = WMMA_F16(af, Bsw[p][nb + 1][lane], acc1);
    if (s < 15) { cvt_store8(adst[1 - p], ra); cvt_store8(bdst[1 - p], rb); }
    __syncthreads();
  }
#pragma unroll
  for (int j = 0; j < 2; ++j) {
    const v8f acc = j ? acc1 : acc0;
#pragma unroll
    for (int r = 0; r < 8; ++r) {
      const int row = m0 + mi * 16 + (lane >> 4) * 8 + r;   // b*L + l
      const int col = n0 + (nb + j) * 16 + (lane & 15);     // h*32 + d
      const int bi = row >> 10, l = row & (Ll - 1);
      const int hh = col >> 5,  d = col & 31;
      const size_t bh = (size_t)(bi * Hh + hh);
      const size_t idx = vt ? (bh * DKk + d) * Ll + l : (bh * Ll + l) * DKk + d;
      out[idx] = (_Float16)(acc[r] * scale);
    }
  }
}

// ---- scores: attn_raw = mask ? Q.K^T : -1e9 (1/sqrt(128) folded into Q) ----
// per (b,h): tile 64 queries x 64 keys, K-dim 32 = one WMMA; all-async f16 staging
__global__ void scores_kernel(const _Float16* __restrict__ Q, const _Float16* __restrict__ Kt,
                              const int* __restrict__ mask, float* __restrict__ attn) {
  __shared__ v16h Qsw[4][32];
  __shared__ v16h Ksw[4][32];
  const int tid = threadIdx.x, wave = tid >> 5, lane = tid & 31;
  const int m0 = blockIdx.x * 64, n0 = blockIdx.y * 64, bh = blockIdx.z;
  const int bi = bh >> 3;
  const _Float16* Qbh = Q  + (size_t)bh * Ll * DKk;
  const _Float16* Kbh = Kt + (size_t)bh * Ll * DKk;
  {
    const int row = tid >> 2, c = tid & 3;
    async_b128(lds_off((_Float16*)&Qsw[row >> 4][(c >> 1) * 16 + (row & 15)] + (c & 1) * 8),
               Qbh + (size_t)(m0 + row) * DKk + c * 8);
    const int lane_k = (c < 2 ? 0 : 16) + (row & 15);
    async_b128(lds_off((_Float16*)&Ksw[row >> 4][lane_k] + (c & 1) * 8),
               Kbh + (size_t)(n0 + row) * DKk + c * 8);
  }
  wait_async0();
  __syncthreads();
  const int mi = wave >> 1, nb = (wave & 1) * 2;
  const v16h af = Qsw[mi][lane];
  v8f acc0 = {}, acc1 = {};
  acc0 = WMMA_F16(af, Ksw[nb][lane], acc0);
  acc1 = WMMA_F16(af, Ksw[nb + 1][lane], acc1);
#pragma unroll
  for (int j = 0; j < 2; ++j) {
    const v8f acc = j ? acc1 : acc0;
#pragma unroll
    for (int r = 0; r < 8; ++r) {
      const int ql = m0 + mi * 16 + (lane >> 4) * 8 + r;
      const int kl = n0 + (nb + j) * 16 + (lane & 15);
      const int mv = mask[(size_t)bi * Ll * Ll + (size_t)ql * Ll + kl];
      attn[((size_t)bh * Ll + ql) * Ll + kl] = mv ? acc[r] : NEGV;
    }
  }
}

// ---------------- row softmax in place over 1024 columns ----------------
__global__ void softmax_kernel(float* __restrict__ attn) {
  __shared__ float red[256];
  const int t = threadIdx.x;
  float* p = attn + (size_t)blockIdx.x * Ll;
  float v[4];
  float mx = -INFINITY;
#pragma unroll
  for (int i = 0; i < 4; ++i) { v[i] = p[t + i * 256]; mx = fmaxf(mx, v[i]); }
  red[t] = mx; __syncthreads();
  for (int s = 128; s > 0; s >>= 1) { if (t < s) red[t] = fmaxf(red[t], red[t + s]); __syncthreads(); }
  mx = red[0]; __syncthreads();
  float sum = 0.f;
#pragma unroll
  for (int i = 0; i < 4; ++i) { v[i] = __expf(v[i] - mx); sum += v[i]; }
  red[t] = sum; __syncthreads();
  for (int s = 128; s > 0; s >>= 1) { if (t < s) red[t] += red[t + s]; __syncthreads(); }
  const float inv = 1.0f / red[0];
#pragma unroll
  for (int i = 0; i < 4; ++i) p[t + i * 256] = v[i] * inv;
}

// ---- AO = attn @ V : per (b,h) [1024,1024]x[1024,32]; V stored transposed [bh][d][l]
// tile 64x32, k-step 32, double-buffered; attn cvt f32->f16, V async b128
__global__ void av_kernel(const float* __restrict__ attn, const _Float16* __restrict__ Vt,
                          _Float16* __restrict__ AO) {
  __shared__ v16h Asw[2][4][32];
  __shared__ v16h Bsw[2][2][32];
  const int tid = threadIdx.x, wave = tid >> 5, lane = tid & 31;
  const int m0 = blockIdx.x * 64, bh = blockIdx.y;
  const int mi = wave >> 1, ni = wave & 1;
  const float* Abh = attn + (size_t)bh * Ll * Ll;
  const _Float16* Vbh = Vt + (size_t)bh * Ll * DKk;  // [d][l]
  const int arow = tid >> 2, akoct = tid & 3;
  _Float16* adst[2];
  unsigned vdst[2];
  const int vd = tid >> 2, vkc = tid & 3;            // valid for tid < 128
#pragma unroll
  for (int p = 0; p < 2; ++p) {
    adst[p] = (_Float16*)&Asw[p][arow >> 4][(akoct >> 1) * 16 + (arow & 15)] + (akoct & 1) * 8;
    vdst[p] = lds_off((_Float16*)&Bsw[p][vd >> 4][(vkc >> 1) * 16 + (vd & 15)] + (vkc & 1) * 8);
  }
  const float* asrc = &Abh[(size_t)(m0 + arow) * Ll + akoct * 8];
  const _Float16* vsrc = Vbh + (size_t)vd * Ll + vkc * 8;
  auto load_a = [&](int k0, float ra[8]) {
    const float4 f0 = *(const float4*)(asrc + k0);
    const float4 f1 = *(const float4*)(asrc + k0 + 4);
    ra[0] = f0.x; ra[1] = f0.y; ra[2] = f0.z; ra[3] = f0.w;
    ra[4] = f1.x; ra[5] = f1.y; ra[6] = f1.z; ra[7] = f1.w;
  };
  {
    float ra[8];
    load_a(0, ra);
    if (tid < 128) async_b128(vdst[0], vsrc);
    cvt_store8(adst[0], ra);
  }
  wait_async0();
  __syncthreads();
  v8f acc = {};
  for (int s = 0; s < 32; ++s) {
    const int p = s & 1;
    float ra[8];
    if (s < 31) {
      if (tid < 128) async_b128(vdst[1 - p], vsrc + (s + 1) * 32);
      load_a((s + 1) * 32, ra);
    }
    acc = WMMA_F16(Asw[p][mi][lane], Bsw[p][ni][lane], acc);
    if (s < 31) cvt_store8(adst[1 - p], ra);
    wait_async0();
    __syncthreads();
  }
  const int bi = bh >> 3, hh = bh & 7;
#pragma unroll
  for (int r = 0; r < 8; ++r) {
    const int ql = m0 + mi * 16 + (lane >> 4) * 8 + r;
    const int d  = ni * 16 + (lane & 15);
    AO[((size_t)(bi * Ll + ql)) * Dd + hh * DKk + d] = (_Float16)acc[r];
  }
}

// ---- Y = AO @ w_fc : [4096,256]x[256,256] -> f32; A async-staged, W converted ----
__global__ void fc_kernel(const _Float16* __restrict__ A, const float* __restrict__ W,
                          float* __restrict__ Y) {
  __shared__ v16h Asw[2][4][32];
  __shared__ v16h Bsw[2][4][32];
  const int tid = threadIdx.x, wave = tid >> 5, lane = tid & 31;
  const int m0 = blockIdx.x * 64, n0 = blockIdx.y * 64;
  const int mi = wave >> 1, nb = (wave & 1) * 2;
  const int arow = tid >> 2, ac = tid & 3;
  const int bcol = tid & 63, bkoct = tid >> 6;
  unsigned adst[2]; _Float16* bdst[2];
#pragma unroll
  for (int p = 0; p < 2; ++p) {
    adst[p] = lds_off((_Float16*)&Asw[p][arow >> 4][(ac >> 1) * 16 + (arow & 15)] + (ac & 1) * 8);
    bdst[p] = (_Float16*)&Bsw[p][bcol >> 4][(bkoct >> 1) * 16 + (bcol & 15)] + (bkoct & 1) * 8;
  }
  const _Float16* asrc = A + (size_t)(m0 + arow) * Dd + ac * 8;
  auto load_w = [&](int k0, float rb[8]) {
#pragma unroll
    for (int j = 0; j < 8; ++j)
      rb[j] = W[(size_t)(k0 + bkoct * 8 + j) * Dd + n0 + bcol];
  };
  {
    float rb[8];
    async_b128(adst[0], asrc);
    load_w(0, rb);
    cvt_store8(bdst[0], rb);
  }
  wait_async0();
  __syncthreads();
  v8f acc0 = {}, acc1 = {};
  for (int s = 0; s < 8; ++s) {
    const int p = s & 1;
    float rb[8];
    if (s < 7) {
      async_b128(adst[1 - p], asrc + (s + 1) * 32);
      load_w((s + 1) * 32, rb);
    }
    const v16h af = Asw[p][mi][lane];
    acc0 = WMMA_F16(af, Bsw[p][nb][lane], acc0);
    acc1 = WMMA_F16(af, Bsw[p][nb + 1][lane], acc1);
    if (s < 7) cvt_store8(bdst[1 - p], rb);
    wait_async0();
    __syncthreads();
  }
#pragma unroll
  for (int j = 0; j < 2; ++j) {
    const v8f acc = j ? acc1 : acc0;
#pragma unroll
    for (int r = 0; r < 8; ++r) {
      const int row = m0 + mi * 16 + (lane >> 4) * 8 + r;
      const int col = n0 + (nb + j) * 16 + (lane & 15);
      Y[(size_t)row * Dd + col] = acc[r];
    }
  }
}

// ---------------- out = LayerNorm(Y + residual) * g + b ----------------
__global__ void ln_kernel(const float* __restrict__ Y, const float* __restrict__ res,
                          const float* __restrict__ g, const float* __restrict__ bta,
                          float* __restrict__ out) {
  __shared__ float red[256];
  const int row = blockIdx.x, t = threadIdx.x;
  const float v = Y[(size_t)row * Dd + t] + res[(size_t)row * Dd + t];
  red[t] = v; __syncthreads();
  for (int s = 128; s > 0; s >>= 1) { if (t < s) red[t] += red[t + s]; __syncthreads(); }
  const float mu = red[0] * (1.0f / Dd); __syncthreads();
  const float dv = v - mu;
  red[t] = dv * dv; __syncthreads();
  for (int s = 128; s > 0; s >>= 1) { if (t < s) red[t] += red[t + s]; __syncthreads(); }
  const float var = red[0] * (1.0f / Dd);
  out[(size_t)row * Dd + t] = dv * rsqrtf(var + EPSV) * g[t] + bta[t];
}

extern "C" void kernel_launch(void* const* d_in, const int* in_sizes, int n_in,
                              void* d_out, int out_size, void* d_ws, size_t ws_size,
                              hipStream_t stream) {
  (void)in_sizes; (void)n_in; (void)out_size; (void)ws_size;
  const float* q_a   = (const float*)d_in[0];
  const float* k_a   = (const float*)d_in[1];
  const float* v_a   = (const float*)d_in[2];
  const float* q_s   = (const float*)d_in[3];
  const float* k_s   = (const float*)d_in[4];
  const float* v_s   = (const float*)d_in[5];
  const float* q_bh  = (const float*)d_in[6];
  const float* q_bah = (const float*)d_in[7];
  const float* q_bbh = (const float*)d_in[8];
  const float* q_pan = (const float*)d_in[9];
  const float* q_oan = (const float*)d_in[10];
  const int*   mask  = (const int*)d_in[11];
  const float* w_qa  = (const float*)d_in[12];
  const float* w_ka  = (const float*)d_in[13];
  const float* w_va  = (const float*)d_in[14];
  const float* w_qs  = (const float*)d_in[15];
  const float* w_ks  = (const float*)d_in[16];
  const float* w_vs  = (const float*)d_in[17];
  const float* w_fc  = (const float*)d_in[18];
  const float* ln_g  = (const float*)d_in[19];
  const float* ln_b  = (const float*)d_in[20];

  float* out  = (float*)d_out;                       // [B,L,D]
  float* attn = out + (size_t)BL * Dd;               // [B,H,L,L]

  // workspace carve-up (~16 MB)
  char* w = (char*)d_ws;
  float*    resF = (float*)w;    w += (size_t)BL * Dd * sizeof(float);
  _Float16* Qh   = (_Float16*)w; w += (size_t)BH * Ll * DKk * sizeof(_Float16);
  _Float16* Kh   = (_Float16*)w; w += (size_t)BH * Ll * DKk * sizeof(_Float16);
  _Float16* Vth  = (_Float16*)w; w += (size_t)BH * Ll * DKk * sizeof(_Float16);
  _Float16* AOh  = (_Float16*)w; w += (size_t)BL * Dd * sizeof(_Float16);
  float*    Yf   = (float*)w;    w += (size_t)BL * Dd * sizeof(float);

  const dim3 blk(256);
  residual_kernel<<<(BL * Dd) / 256, blk, 0, stream>>>(q_a, q_s, q_bh, q_bah, q_bbh,
                                                       q_pan, q_oan, resF);
  proj_kernel<<<dim3(BL / 64, Dd / 64), blk, 0, stream>>>(q_a, w_qa, q_s, w_qs, Qh, QSCALE, 0);
  proj_kernel<<<dim3(BL / 64, Dd / 64), blk, 0, stream>>>(k_a, w_ka, k_s, w_ks, Kh, 1.0f, 0);
  proj_kernel<<<dim3(BL / 64, Dd / 64), blk, 0, stream>>>(v_a, w_va, v_s, w_vs, Vth, 1.0f, 1);
  scores_kernel<<<dim3(Ll / 64, Ll / 64, BH), blk, 0, stream>>>(Qh, Kh, mask, attn);
  softmax_kernel<<<BH * Ll, blk, 0, stream>>>(attn);
  av_kernel<<<dim3(Ll / 64, BH), blk, 0, stream>>>(attn, Vth, AOh);
  fc_kernel<<<dim3(BL / 64, Dd / 64), blk, 0, stream>>>(AOh, w_fc, Yf);
  ln_kernel<<<BL, blk, 0, stream>>>(Yf, resF, ln_g, ln_b, out);
}
